// CustomMultiheadAttention_76355928588396
// MI455X (gfx1250) — matrix operand order repaired and checked by reading, not compiled, not run on hardware
//
#include <hip/hip_runtime.h>

// ---------------------------------------------------------------------------
// CDNA5 (gfx1250) fused multi-head attention, bf16 WMMA with f32 accumulate.
// Weight panels staged to LDS via the Tensor Data Mover (tensor_load_to_lds).
// ---------------------------------------------------------------------------

typedef __bf16 bf16;
typedef __attribute__((ext_vector_type(16))) __bf16 v16bf;
typedef __attribute__((ext_vector_type(8)))  __bf16 v8bf;
typedef __attribute__((ext_vector_type(4)))  __bf16 v4bf;
typedef __attribute__((ext_vector_type(8)))  float  v8f;
typedef __attribute__((ext_vector_type(4)))  unsigned int u32x4;
typedef __attribute__((ext_vector_type(4)))  int i32x4;
typedef __attribute__((ext_vector_type(8)))  int i32x8;

#define DIM   1024
#define NH    16
#define HD    64
#define BATCH 4
#define SQL   2048
#define SKL   2048
#define MROWS (BATCH * SQL)   // 8192

// ---------------------------------------------------------------------------
// Per-lane operand loaders matching the ISA 16-bit WMMA VGPR layouts.
// A matrix 16x32 (MxK): lane = row (M = lane&15); lanes<16 hold K {0..7,16..23},
// lanes>=16 hold K {8..15,24..31}; 8 contiguous bf16 per half -> b128 loads.
// ---------------------------------------------------------------------------
__device__ __forceinline__ v16bf load_a16x32(const bf16* __restrict__ base,
                                             int ld, int lane) {
  const int row = lane & 15;
  const int kh  = (lane & 16) ? 8 : 0;
  const bf16* p = base + (long)row * ld + kh;
  v8bf x0 = *(const v8bf*)p;
  v8bf x1 = *(const v8bf*)(p + 16);
  v16bf r;
#pragma unroll
  for (int i = 0; i < 8; ++i) { r[i] = x0[i]; r[i + 8] = x1[i]; }
  return r;
}

// B matrix 32x16 (KxN) with B[k][n] = R[n][k] (R row-major, row length ld):
// lane = column (N = lane&15); lanes<16 hold K {0..7,16..23}, lanes>=16 hold K {8..15,24..31}.
__device__ __forceinline__ v16bf load_b32x16(const bf16* __restrict__ base,
                                             int ld, int lane) {
  const int n  = lane & 15;
  const int kh = (lane & 16) ? 8 : 0;
  const bf16* p = base + (long)n * ld + kh;
  v8bf x0 = *(const v8bf*)p;
  v8bf x1 = *(const v8bf*)(p + 16);
  v16bf r;
#pragma unroll
  for (int i = 0; i < 8; ++i) { r[i] = x0[i]; r[i + 8] = x1[i]; }
  return r;
}

__device__ __forceinline__ v8f wmma_bf16(v16bf a, v16bf b, v8f c) {
  return __builtin_amdgcn_wmma_f32_16x16x32_bf16(
      false, a, false, b, (short)0, c, false, false);
}

// ---------------------------------------------------------------------------
// Tensor Data Mover: load a 64-row x 128-K bf16 tile of W (row stride 1024
// elems) into LDS at lds_off. D# built per ISA ch.8 bit layout.
// ---------------------------------------------------------------------------
__device__ __forceinline__ void tdm_load_w_panel(const bf16* gaddr,
                                                 unsigned lds_off) {
  const unsigned long long ga = (unsigned long long)gaddr;
  u32x4 g0;
  g0[0] = 1u;                                    // count=1, user descriptor
  g0[1] = lds_off;                               // lds_addr (bytes)
  g0[2] = (unsigned)(ga & 0xffffffffu);          // global_addr[31:0]
  g0[3] = (unsigned)((ga >> 32) & 0x01ffffffu)   // global_addr[56:32]
        | (2u << 30);                            // type = 2 ("image")
  i32x8 g1;
  g1[0] = (1 << 16);                             // data_size = 1 (2 bytes)
  g1[1] = (int)(1024u << 16);                    // tensor_dim0 = 1024 (lo16)
  g1[2] = (int)(1024u << 16);                    // dim0 hi=0 | tensor_dim1 lo16 = 1024
  g1[3] = (int)(128u  << 16);                    // dim1 hi=0 | tile_dim0 = 128
  g1[4] = 64;                                    // tile_dim1 = 64, tile_dim2 = 0
  g1[5] = 1024;                                  // tensor_dim0_stride = 1024
  g1[6] = 0;                                     // stride hi | dim1_stride lo
  g1[7] = 0;
  const i32x4 z4 = {0, 0, 0, 0};
#if defined(__clang_major__) && (__clang_major__ >= 23)
  const i32x8 z8 = {0, 0, 0, 0, 0, 0, 0, 0};
  __builtin_amdgcn_tensor_load_to_lds(g0, g1, z4, z4, z8, 0);
#else
  __builtin_amdgcn_tensor_load_to_lds(g0, g1, z4, z4, 0);
#endif
}

// ---------------------------------------------------------------------------
// f32 -> bf16 pack (vectorized by 4)
// ---------------------------------------------------------------------------
__global__ void __launch_bounds__(256)
cvt_f32_bf16(const float4* __restrict__ in, v4bf* __restrict__ out, long n4) {
  long i = blockIdx.x * (long)blockDim.x + threadIdx.x;
  const long stride = (long)gridDim.x * blockDim.x;
  for (; i < n4; i += stride) {
    float4 t = in[i];
    v4bf o;
    o[0] = (bf16)t.x; o[1] = (bf16)t.y; o[2] = (bf16)t.z; o[3] = (bf16)t.w;
    out[i] = o;
  }
}

// ---------------------------------------------------------------------------
// GEMM: out = X[M,1024] @ W^T + bias.  Block tile 128M x 64N, 8 waves,
// wave tile 32x32 (4 accumulators, 2 loads/WMMA).  Weight panel (64x128 bf16,
// 16KB) staged to LDS by the TDM, double-buffered, s_wait_tensorcnt synced.
// mode 0: out bf16 [B,H,S,64] | mode 1: out bf16 [B,H,64,S] | mode 2: f32 [M,N]
// ---------------------------------------------------------------------------
__global__ void __launch_bounds__(256)
gemm_bf16(const bf16* __restrict__ X, const bf16* __restrict__ W,
          const float* __restrict__ bias, void* __restrict__ out, int mode) {
  __shared__ bf16 Bs[2][64][128];                // 2 x 16 KB

  const int lane = threadIdx.x & 31;
  const int wave = threadIdx.x >> 5;
  const int bm = blockIdx.x >> 4;                // 64 M-blocks
  const int bn = blockIdx.x & 15;                // 16 N-blocks
  const int m0 = bm * 128 + (wave >> 1) * 32;    // wave rows
  const int n0 = bn * 64  + (wave & 1) * 32;     // wave cols (global)
  const int ln0 = (wave & 1) * 32;               // wave cols (LDS-local)

  const int col   = lane & 15;
  const int rbase = (lane & 16) ? 8 : 0;
  const int bkh   = (lane & 16) ? 8 : 0;         // B-operand K offset

  v8f acc[2][2];
#pragma unroll
  for (int mi = 0; mi < 2; ++mi)
#pragma unroll
    for (int ni = 0; ni < 2; ++ni) {
      const float bv = bias[n0 + ni * 16 + col];
#pragma unroll
      for (int r = 0; r < 8; ++r) acc[mi][ni][r] = bv;
    }

  const bf16* Abase = X + (long)m0 * DIM;
  const bf16* Wpanel = W + (long)(bn * 64) * DIM;   // 64 rows of W

  // prologue: TDM chunk 0 into buffer 0
  if (wave == 0) {
    tdm_load_w_panel(Wpanel, (unsigned)(unsigned long long)&Bs[0][0][0]);
    __builtin_amdgcn_s_wait_tensorcnt(0);
  }
  __syncthreads();

  for (int chunk = 0; chunk < 8; ++chunk) {       // K chunks of 128
    const int cur = chunk & 1;
    if (wave == 0 && chunk + 1 < 8)
      tdm_load_w_panel(Wpanel + (chunk + 1) * 128,
                       (unsigned)(unsigned long long)&Bs[cur ^ 1][0][0]);
#pragma unroll
    for (int ks = 0; ks < 4; ++ks) {              // 4 k-steps of 32
      const int kglob = chunk * 128 + ks * 32;
      __builtin_prefetch(Abase + kglob + 256, 0, 0);
      v16bf a0 = load_a16x32(Abase + kglob,            DIM, lane);
      v16bf a1 = load_a16x32(Abase + 16 * DIM + kglob, DIM, lane);
      v16bf b0, b1;
#pragma unroll
      for (int ni = 0; ni < 2; ++ni) {
        const bf16* bp = &Bs[cur][ln0 + ni * 16 + col][ks * 32 + bkh];
        v8bf x0 = *(const v8bf*)bp;
        v8bf x1 = *(const v8bf*)(bp + 16);
        v16bf t;
#pragma unroll
        for (int i = 0; i < 8; ++i) { t[i] = x0[i]; t[i + 8] = x1[i]; }
        if (ni == 0) b0 = t; else b1 = t;
      }
      acc[0][0] = wmma_bf16(a0, b0, acc[0][0]);
      acc[0][1] = wmma_bf16(a0, b1, acc[0][1]);
      acc[1][0] = wmma_bf16(a1, b0, acc[1][0]);
      acc[1][1] = wmma_bf16(a1, b1, acc[1][1]);
    }
    if (wave == 0 && chunk + 1 < 8) __builtin_amdgcn_s_wait_tensorcnt(0);
    __syncthreads();
  }

#pragma unroll
  for (int mi = 0; mi < 2; ++mi)
#pragma unroll
    for (int ni = 0; ni < 2; ++ni) {
      const int mt = m0 + mi * 16;
      const int nt = n0 + ni * 16;
      if (mode == 2) {
        float* O = (float*)out;
#pragma unroll
        for (int r = 0; r < 8; ++r)
          O[(long)(mt + rbase + r) * DIM + nt + col] = acc[mi][ni][r];
      } else {
        bf16* O = (bf16*)out;
        const int h  = (nt + col) >> 6;
        const int hd = (nt + col) & 63;
#pragma unroll
        for (int r = 0; r < 8; ++r) {
          const int m  = mt + rbase + r;          // m = b*SQL + s
          const int bb = m >> 11;
          const int s  = m & (SQL - 1);
          long idx;
          if (mode == 0) idx = (((long)(bb * NH + h) * SQL) + s) * HD + hd;
          else           idx = (((long)(bb * NH + h) * HD)  + hd) * SKL + s;
          O[idx] = (bf16)acc[mi][ni][r];
        }
      }
    }
}

// ---------------------------------------------------------------------------
// Attention: one workgroup (8 waves) per (b,h, 16 query rows).
//   S = scale * Q K^T -> 128KB LDS f32 ; softmax (wave reductions) ->
//   probs to d_out f32 + bf16 LDS copy ; O = P @ V with all 8 waves
//   (4 head-dim slices x 2 K-halves, partial-sum reduce through LDS).
// ---------------------------------------------------------------------------
__global__ void __launch_bounds__(256)
attn_kernel(const bf16* __restrict__ Qh, const bf16* __restrict__ Kh,
            const bf16* __restrict__ Vt, bf16* __restrict__ Ctx,
            float* __restrict__ attn_out) {
  __shared__ float Sf[16][SKL];                   // 128 KB
  __shared__ bf16  Pb[16][SKL];                   //  64 KB

  const int lane = threadIdx.x & 31;
  const int wave = threadIdx.x >> 5;
  const int qt = blockIdx.x & (SQL / 16 - 1);
  const int bh = blockIdx.x >> 7;                 // b*NH + h
  const float scale = 0.125f;                     // 64^-0.5

  const int col   = lane & 15;
  const int rbase = (lane & 16) ? 8 : 0;

  // Q tile (16x64) in registers: two A operands (K=0..31, 32..63)
  const bf16* Qbase = Qh + ((long)bh * SQL + qt * 16) * HD;
  v16bf aq0 = load_a16x32(Qbase,      HD, lane);
  v16bf aq1 = load_a16x32(Qbase + 32, HD, lane);

  // ---- scores: waves stripe the 128 key tiles ----
  for (int j = wave; j < SKL / 16; j += 8) {
    const bf16* Kbase = Kh + ((long)bh * SKL + j * 16) * HD;
    __builtin_prefetch(Kbase + 8 * 16 * HD, 0, 0);
    v16bf bk0 = load_b32x16(Kbase,      HD, lane);
    v16bf bk1 = load_b32x16(Kbase + 32, HD, lane);
    v8f c = 0.0f;
    c = wmma_bf16(aq0, bk0, c);
    c = wmma_bf16(aq1, bk1, c);
#pragma unroll
    for (int r = 0; r < 8; ++r)
      Sf[rbase + r][j * 16 + col] = c[r] * scale;
  }
  __syncthreads();

  // ---- softmax: each wave owns 2 rows ----
#pragma unroll
  for (int rr = 0; rr < 2; ++rr) {
    const int row = wave * 2 + rr;
    float m = -3.0e38f;
    for (int ci = lane; ci < SKL; ci += 32) m = fmaxf(m, Sf[row][ci]);
#pragma unroll
    for (int off = 16; off > 0; off >>= 1) m = fmaxf(m, __shfl_xor(m, off, 32));
    float s = 0.0f;
    for (int ci = lane; ci < SKL; ci += 32) s += __expf(Sf[row][ci] - m);
#pragma unroll
    for (int off = 16; off > 0; off >>= 1) s += __shfl_xor(s, off, 32);
    const float inv = 1.0f / s;
    float* orow = attn_out + ((long)bh * SQL + qt * 16 + row) * SKL;
    for (int ci = lane; ci < SKL; ci += 32) {
      const float p = __expf(Sf[row][ci] - m) * inv;
      orow[ci] = p;
      Pb[row][ci] = (bf16)p;
    }
  }
  __syncthreads();

  // ---- O = P @ V: wave -> (head slice nt, K half) ----
  {
    const int nt    = wave & 3;
    const int khalf = wave >> 2;
    const int kbeg  = khalf * (SKL / 2);
    const bf16* Vbase = Vt + ((long)bh * HD + nt * 16) * SKL + kbeg;
    v8f c = 0.0f;
    const int arow = lane & 15;
    const int akh  = (lane & 16) ? 8 : 0;
    for (int k0 = 0; k0 < SKL / 2; k0 += 32) {
      v8bf x0 = *(const v8bf*)&Pb[arow][kbeg + k0 + akh];
      v8bf x1 = *(const v8bf*)&Pb[arow][kbeg + k0 + akh + 16];
      v16bf a;
#pragma unroll
      for (int i = 0; i < 8; ++i) { a[i] = x0[i]; a[i + 8] = x1[i]; }
      v16bf b = load_b32x16(Vbase + k0, SKL, lane);
      c = wmma_bf16(a, b, c);
    }

    // reduce the two K-halves through LDS (reuse score storage)
    float* red = &Sf[0][0];                       // 16 rows x 64 cols
    if (khalf == 1) {
#pragma unroll
      for (int r = 0; r < 8; ++r)
        red[(rbase + r) * HD + nt * 16 + col] = c[r];
    }
    __syncthreads();
    if (khalf == 0) {
      const int bq = bh >> 4;
      const int h  = bh & 15;
#pragma unroll
      for (int r = 0; r < 8; ++r) {
        const float v2 = c[r] + red[(rbase + r) * HD + nt * 16 + col];
        const int s = qt * 16 + rbase + r;
        Ctx[((long)bq * SQL + s) * DIM + h * HD + nt * 16 + col] = (bf16)v2;
      }
    }
  }
}

// ---------------------------------------------------------------------------
// Launch
// ---------------------------------------------------------------------------
extern "C" void kernel_launch(void* const* d_in, const int* in_sizes, int n_in,
                              void* d_out, int out_size, void* d_ws, size_t ws_size,
                              hipStream_t stream) {
  const float* q  = (const float*)d_in[0];
  const float* k  = (const float*)d_in[1];
  const float* v  = (const float*)d_in[2];
  const float* Wq = (const float*)d_in[3];
  const float* bq = (const float*)d_in[4];
  const float* Wk = (const float*)d_in[5];
  const float* bk = (const float*)d_in[6];
  const float* Wv = (const float*)d_in[7];
  const float* bv = (const float*)d_in[8];
  const float* Wo = (const float*)d_in[9];
  const float* bo = (const float*)d_in[10];

  const long NX = (long)MROWS * DIM;    // 8,388,608 activations per tensor
  const long NW = (long)DIM * DIM;      // 1,048,576 per weight

  char* ws = (char*)d_ws;
  bf16* Xq  = (bf16*)ws;                 ws += NX * 2;
  bf16* Xk  = (bf16*)ws;                 ws += NX * 2;
  bf16* Xv  = (bf16*)ws;                 ws += NX * 2;
  bf16* WqB = (bf16*)ws;                 ws += NW * 2;
  bf16* WkB = (bf16*)ws;                 ws += NW * 2;
  bf16* WvB = (bf16*)ws;                 ws += NW * 2;
  bf16* WoB = (bf16*)ws;                 ws += NW * 2;
  bf16* Qh  = (bf16*)ws;                 ws += NX * 2;
  bf16* Kh  = (bf16*)ws;                 ws += NX * 2;
  bf16* Vt  = (bf16*)ws;                 ws += NX * 2;
  bf16* Ctx = (bf16*)ws;                 ws += NX * 2;

  float* out_x    = (float*)d_out;
  float* out_attn = out_x + NX;

  // 1) f32 -> bf16 conversions
  cvt_f32_bf16<<<8192, 256, 0, stream>>>((const float4*)q,  (v4bf*)Xq,  NX / 4);
  cvt_f32_bf16<<<8192, 256, 0, stream>>>((const float4*)k,  (v4bf*)Xk,  NX / 4);
  cvt_f32_bf16<<<8192, 256, 0, stream>>>((const float4*)v,  (v4bf*)Xv,  NX / 4);
  cvt_f32_bf16<<<1024, 256, 0, stream>>>((const float4*)Wq, (v4bf*)WqB, NW / 4);
  cvt_f32_bf16<<<1024, 256, 0, stream>>>((const float4*)Wk, (v4bf*)WkB, NW / 4);
  cvt_f32_bf16<<<1024, 256, 0, stream>>>((const float4*)Wv, (v4bf*)WvB, NW / 4);
  cvt_f32_bf16<<<1024, 256, 0, stream>>>((const float4*)Wo, (v4bf*)WoB, NW / 4);

  // 2) projections: (8192/128) * (1024/64) = 1024 blocks
  gemm_bf16<<<1024, 256, 0, stream>>>(Xq, WqB, bq, (void*)Qh, 0);
  gemm_bf16<<<1024, 256, 0, stream>>>(Xk, WkB, bk, (void*)Kh, 0);
  gemm_bf16<<<1024, 256, 0, stream>>>(Xv, WvB, bv, (void*)Vt, 1);

  // 3) attention: B*H*(SQ/16) = 8192 workgroups
  attn_kernel<<<BATCH * NH * (SQL / 16), 256, 0, stream>>>(Qh, Kh, Vt, Ctx, out_attn);

  // 4) output projection -> x (f32)
  gemm_bf16<<<1024, 256, 0, stream>>>(Ctx, WoB, bo, (void*)out_x, 2);
}
